// CGRNN_batch_Var_33741263078255
// MI455X (gfx1250) — matrix operand order repaired and checked by reading, not compiled, not run on hardware
//
#include <hip/hip_runtime.h>
#include <hip/hip_bf16.h>
#include <math.h>

typedef __attribute__((ext_vector_type(16))) _Float16 v16h;
typedef __attribute__((ext_vector_type(8)))  float    v8f;

#define BB 32
#define TT 128
#define NN 128
#define DD 64
#define FF 129      // 2D+1
#define FPAD 144    // padded cols for adjacency GEMM (9 x 16)
#define KP 160      // padded K for gate GEMMs (5 x 32)
#define AS 132      // sh_adj stride (halves)
#define XS 148      // sh_x stride (halves)
#define NBLK 64

// ---- workspace layout (bytes) ----
#define OFF_GN    0u
#define OFF_ADJS  4096u
#define OFF_VTO   69632u
#define OFF_BAR   86016u
#define OFF_H     86272u                      // 32*128*64*4      = 1048576
#define OFF_COMB  (OFF_H   + 1048576u)        // 128*32*160*2     = 1310720
#define OFF_XCAT  (OFF_COMB + 1310720u)       // 128*32*160*2     = 1310720
#define OFF_WR    (OFF_XCAT + 1310720u)       // 128*160*64*2     = 2621440
#define OFF_WU    (OFF_WR  + 2621440u)
#define OFF_WC    (OFF_WU  + 2621440u)

__device__ __forceinline__ float sigmf(float x) { return 1.f / (1.f + expf(-x)); }

__device__ __forceinline__ v16h load_afrag(const _Float16* base, int stride,
                                           int row0, int k0, int lane) {
    const int half = lane >> 4, lm = lane & 15;
    const _Float16* p = base + (row0 + lm) * stride + k0;
    v16h a;
#pragma unroll
    for (int e = 0; e < 16; ++e) {
        const int v  = e >> 1;
        const int kk = ((v & 4) << 2) + half * 8 + (v & 3) * 2 + (e & 1);
        a[e] = p[kk];
    }
    return a;
}

__device__ __forceinline__ v16h load_bfrag(const _Float16* base, int stride,
                                           int k0, int col0, int lane) {
    const int half = lane >> 4, lm = lane & 15;
    const _Float16* p = base + k0 * stride + col0 + lm;
    v16h b;
#pragma unroll
    for (int e = 0; e < 16; ++e) {
        const int v  = e >> 1;
        const int kk = ((v & 4) << 2) + half * 8 + (v & 3) * 2 + (e & 1);
        b[e] = p[kk * stride];
    }
    return b;
}

#define WMMA(A, B, C) __builtin_amdgcn_wmma_f32_16x16x32_f16(false, (A), false, (B), (short)0, (C), false, false)

// ---------------- device-wide barrier (persistent grid of NBLK blocks) ----------------
__device__ __forceinline__ void gsync(unsigned* cnt, volatile unsigned* gen) {
    __syncthreads();
    if (threadIdx.x == 0) {
        __threadfence();
        const unsigned g = *gen;
        const unsigned a = atomicAdd(cnt, 1u);
        if (a == NBLK - 1u) {
            *cnt = 0u;
            __threadfence();
            *gen = g + 1u;
        } else {
            while (*gen == g) __builtin_amdgcn_s_sleep(2);
        }
        __threadfence();
    }
    __syncthreads();
}

// ---------------- Precompute: gn ----------------
__global__ __launch_bounds__(128) void k_gn(const float* __restrict__ vp,
                                            const float* __restrict__ W1,
                                            const float* __restrict__ b1,
                                            const float* __restrict__ W2,
                                            const float* __restrict__ b2,
                                            float* __restrict__ gn) {
    __shared__ float vrow[768];
    __shared__ float gh[128];
    __shared__ float gv[8];
    __shared__ float nrm;
    const int n = blockIdx.x, tid = threadIdx.x;
    for (int k = tid; k < 768; k += 128) vrow[k] = vp[n * 768 + k];
    __syncthreads();
    float acc = b1[tid];
    for (int k = 0; k < 768; ++k) acc += vrow[k] * W1[k * 128 + tid];
    gh[tid] = fmaxf(acc, 0.f);
    __syncthreads();
    if (tid < 8) {
        float a = b2[tid];
        for (int k = 0; k < 128; ++k) a += gh[k] * W2[k * 8 + tid];
        gv[tid] = a;
    }
    __syncthreads();
    if (tid == 0) {
        float s = 0.f;
        for (int g = 0; g < 8; ++g) s += gv[g] * gv[g];
        nrm = fmaxf(sqrtf(s), 1e-12f);
    }
    __syncthreads();
    if (tid < 8) gn[n * 8 + tid] = gv[tid] / nrm;
}

// ---------------- Precompute: adj_s = softmax(gn @ gn^T) ----------------
__global__ __launch_bounds__(128) void k_adjs(const float* __restrict__ gn,
                                              float* __restrict__ adjs) {
    __shared__ float gi[8];
    __shared__ float red[128];
    const int i = blockIdx.x, j = threadIdx.x;
    if (j < 8) gi[j] = gn[i * 8 + j];
    __syncthreads();
    float s = 0.f;
    for (int g = 0; g < 8; ++g) s += gi[g] * gn[j * 8 + g];
    red[j] = s;
    __syncthreads();
    for (int off = 64; off > 0; off >>= 1) {
        if (j < off) red[j] = fmaxf(red[j], red[j + off]);
        __syncthreads();
    }
    const float mx = red[0];
    __syncthreads();
    const float e = expf(s - mx);
    red[j] = e;
    __syncthreads();
    for (int off = 64; off > 0; off >>= 1) {
        if (j < off) red[j] += red[j + off];
        __syncthreads();
    }
    adjs[i * 128 + j] = e / red[0];
}

// ---------------- Precompute: var_total_obs ----------------
__global__ __launch_bounds__(256) void k_vto(const float* __restrict__ mask,
                                             float* __restrict__ vto) {
    const int idx = blockIdx.x * blockDim.x + threadIdx.x;
    if (idx >= BB * NN) return;
    const int b = idx >> 7, n = idx & 127;
    float s = 0.f;
    for (int t = 0; t < TT; ++t) s += mask[(b * TT + t) * NN + n];
    vto[idx] = s;
}

// ---------------- Precompute: weights -> f16, K padded 129 -> 160 ----------------
__global__ __launch_bounds__(256) void k_wcvt(const float* __restrict__ Wr,
                                              const float* __restrict__ Wu,
                                              const float* __restrict__ Wc,
                                              _Float16* __restrict__ W16r,
                                              _Float16* __restrict__ W16u,
                                              _Float16* __restrict__ W16c) {
    const int idx = blockIdx.x * 256 + threadIdx.x;     // over 128*160*64
    if (idx >= NN * KP * DD) return;
    const int o = idx & 63;
    const int k = (idx >> 6) % KP;
    const int n = idx / (KP * DD);
    float vr = 0.f, vu = 0.f, vc = 0.f;
    if (k < FF) {
        const int s = (n * FF + k) * DD + o;
        vr = Wr[s]; vu = Wu[s]; vc = Wc[s];
    }
    W16r[idx] = (_Float16)vr;
    W16u[idx] = (_Float16)vu;
    W16c[idx] = (_Float16)vc;
}

// ---------------- Init: zero h, comb16, xcat16, barrier ----------------
__global__ __launch_bounds__(256) void k_init(unsigned* __restrict__ zbase,
                                              unsigned* __restrict__ bar) {
    const int idx = blockIdx.x * 256 + threadIdx.x;
    const int nwords = (1048576 + 1310720 + 1310720) / 4;   // h + comb16 + xcat16
    if (idx < nwords) zbase[idx] = 0u;
    if (blockIdx.x == 0 && threadIdx.x < 2) bar[threadIdx.x] = 0u;
}

// ---------------- Persistent scan: 64 blocks, phase-alternating ----------------
__global__ __launch_bounds__(256) void k_scan(const float* __restrict__ obs,
                                              const float* __restrict__ mask,
                                              const int*   __restrict__ lengths,
                                              const float* __restrict__ avg,
                                              const float* __restrict__ rW,
                                              const float* __restrict__ br_,
                                              const float* __restrict__ bu_,
                                              const float* __restrict__ bc_,
                                              const float* __restrict__ adjs,
                                              const float* __restrict__ vto,
                                              float* __restrict__ hbuf,
                                              _Float16* __restrict__ comb16,
                                              _Float16* __restrict__ xcat16,
                                              const _Float16* __restrict__ W16r,
                                              const _Float16* __restrict__ W16u,
                                              const _Float16* __restrict__ W16c,
                                              unsigned* __restrict__ bar,
                                              float* __restrict__ out) {
    __shared__ _Float16 sh_adj[NN * AS];
    __shared__ _Float16 sh_x[NN * XS];
    __shared__ float sh_obs[NN * DD];      // async-copied obs tile (f32)
    __shared__ float sh_rs[NN];
    __shared__ float sh_m[NN];

    const int tid  = threadIdx.x;
    const int lane = tid & 31;
    const int wv   = tid >> 5;
    const int half = lane >> 4;
    const int lm   = lane & 15;

    // Phase A identity: (batch, tile-half)
    const int b     = blockIdx.x >> 1;
    const int halfb = blockIdx.x & 1;
    // Phase B identity: one item per wave: (node, b-tile, out-half)
    const int gw = blockIdx.x * 8 + wv;          // 0..511
    const int n  = gw >> 2;
    const int mt = (gw >> 1) & 1;
    const int nh = gw & 1;
    const int o0 = nh * 32;
    const int rowA  = n * 32 + mt * 16;          // base row into comb16/xcat16
    const int bRow0 = mt * 16 + half * 8;        // base batch row in D fragment

    unsigned* cnt = bar;
    volatile unsigned* gen = bar + 1;

    for (int t = 0; t < TT; ++t) {
        // ======== Phase A: async-stage obs tile, adjacency build, combined GEMM ====
        {   // kick off 32 KB obs tile -> LDS via async B128 copies (ASYNCcnt)
            const float* srcp = obs + ((size_t)(b * TT + t)) * (NN * DD);
            for (int c = tid; c < (NN * DD) / 4; c += 256) {
                const unsigned ldsoff = (unsigned)(size_t)&sh_obs[c * 4];
                const unsigned goff   = (unsigned)(c * 16);
                asm volatile("global_load_async_to_lds_b128 %0, %1, %2"
                             :: "v"(ldsoff), "v"(goff), "s"(srcp)
                             : "memory");
            }
        }
        if (tid < NN) {
            const float cm = mask[(b * TT + t) * NN + tid];
            const float av = avg[(b * TT + t) * NN + tid];
            sh_m[tid]  = cm;
            sh_rs[tid] = 0.5f * tanhf(av / (vto[b * NN + tid] + 1.f));
        }
        __syncthreads();
        // adjacency build overlaps with the in-flight async copies
        for (int idx = tid; idx < NN * NN; idx += 256) {
            const int i = idx >> 7, j = idx & 127;
            float v;
            if (i == j) v = 1.f;
            else {
                const float rsm = -rW[idx] * fabsf(sh_rs[i] - sh_rs[j]);
                v = adjs[idx] * (1.f + rsm) * sh_m[i] * sh_m[j];
            }
            sh_adj[i * AS + j] = (_Float16)v;
        }
        asm volatile("s_wait_asynccnt 0x0" ::: "memory");
        __syncthreads();
        for (int idx = tid; idx < NN * FPAD; idx += 256) {
            const int nr = idx / FPAD, c = idx % FPAD;
            float v;
            if (c < DD)       v = sh_obs[nr * DD + c];
            else if (c == DD) v = sh_rs[nr];
            else if (c < FF)  v = hbuf[(b * NN + nr) * DD + (c - 65)];
            else              v = 0.f;
            sh_x[nr * XS + c] = (_Float16)v;
        }
        __syncthreads();
        if (halfb == 0) {  // publish x-part [obs|rs] of xcat for the cand GEMM
            for (int idx = tid; idx < NN * 65; idx += 256) {
                const int nr = idx / 65, k = idx % 65;
                xcat16[(nr * 32 + b) * KP + k] = sh_x[nr * XS + k];
            }
        }
        for (int tile = 36 * halfb + wv; tile < 36 * (halfb + 1); tile += 8) {
            const int tm = tile / 9, tn = tile % 9;
            v8f acc = {};
#pragma unroll
            for (int kt = 0; kt < 4; ++kt) {
                v16h a  = load_afrag(sh_adj, AS, tm * 16, kt * 32, lane);
                v16h bm = load_bfrag(sh_x, XS, kt * 32, tn * 16, lane);
                acc = WMMA(a, bm, acc);
            }
            const int col = tn * 16 + lm;
#pragma unroll
            for (int v = 0; v < 8; ++v) {
                const int nrow = tm * 16 + half * 8 + v;
                comb16[(nrow * 32 + b) * KP + col] = (_Float16)acc[v];
            }
        }
        gsync(cnt, gen);

        // ======== Phase B pass 1: r/u gate GEMMs (per-node weights, M = batch) ========
        float hr[2][8], uu[2][8], mm[8];
        {
            const _Float16* WRn = W16r + n * KP * DD;
            const _Float16* WUn = W16u + n * KP * DD;
            __builtin_prefetch(W16c + n * KP * DD + lane, 0, 1);
            v8f ar0 = {}, ar1 = {}, au0 = {}, au1 = {};
#pragma unroll
            for (int kt = 0; kt < 5; ++kt) {
                const int k0 = kt * 32;
                v16h a  = load_afrag(comb16, KP, rowA, k0, lane);
                v16h r0 = load_bfrag(WRn, DD, k0, o0, lane);
                v16h r1 = load_bfrag(WRn, DD, k0, o0 + 16, lane);
                v16h u0 = load_bfrag(WUn, DD, k0, o0, lane);
                v16h u1 = load_bfrag(WUn, DD, k0, o0 + 16, lane);
                ar0 = WMMA(a, r0, ar0);  ar1 = WMMA(a, r1, ar1);
                au0 = WMMA(a, u0, au0);  au1 = WMMA(a, u1, au1);
            }
#pragma unroll
            for (int v = 0; v < 8; ++v)
                mm[v] = mask[((bRow0 + v) * TT + t) * NN + n];
#pragma unroll
            for (int j = 0; j < 2; ++j) {
                const int o = o0 + j * 16 + lm;
                const float biasr = br_[n * DD + o];
                const float biasu = bu_[n * DD + o];
#pragma unroll
                for (int v = 0; v < 8; ++v) {
                    const int brow = bRow0 + v;
                    const float arv = j ? ar1[v] : ar0[v];
                    const float auv = j ? au1[v] : au0[v];
                    const float hv  = hbuf[(brow * NN + n) * DD + o];
                    const float rj  = sigmf(arv + biasr);
                    uu[j][v] = sigmf(auv + biasu);
                    const float hrv = (mm[v] > 0.f) ? rj * hv : hv;
                    hr[j][v] = hrv;
                    xcat16[(n * 32 + brow) * KP + 65 + o] = (_Float16)hrv;
                }
            }
        }
        gsync(cnt, gen);

        // ======== Phase B pass 2: candidate GEMM + GRU update ========
        {
            const _Float16* WCn = W16c + n * KP * DD;
            v8f ac0 = {}, ac1 = {};
#pragma unroll
            for (int kt = 0; kt < 5; ++kt) {
                const int k0 = kt * 32;
                v16h a  = load_afrag(xcat16, KP, rowA, k0, lane);
                v16h c0 = load_bfrag(WCn, DD, k0, o0, lane);
                v16h c1 = load_bfrag(WCn, DD, k0, o0 + 16, lane);
                ac0 = WMMA(a, c0, ac0);  ac1 = WMMA(a, c1, ac1);
            }
#pragma unroll
            for (int j = 0; j < 2; ++j) {
                const int o = o0 + j * 16 + lm;
                const float biasc = bc_[n * DD + o];
#pragma unroll
                for (int v = 0; v < 8; ++v) {
                    const int brow = bRow0 + v;
                    const float acv  = j ? ac1[v] : ac0[v];
                    const float cand = tanhf(acv + biasc);
                    const float hn = (mm[v] > 0.f)
                        ? (1.f - uu[j][v]) * hr[j][v] + uu[j][v] * cand
                        : hr[j][v];
                    hbuf[(brow * NN + n) * DD + o] = hn;
                    if (t == lengths[brow] - 1)
                        out[(brow * NN + n) * DD + o] = hn;
                }
            }
        }
        gsync(cnt, gen);
    }
}

extern "C" void kernel_launch(void* const* d_in, const int* in_sizes, int n_in,
                              void* d_out, int out_size, void* d_ws, size_t ws_size,
                              hipStream_t stream) {
    const float* obs  = (const float*)d_in[0];
    const float* mask = (const float*)d_in[2];
    const int*   len  = (const int*)d_in[5];
    const float* avg  = (const float*)d_in[6];
    const float* vp   = (const float*)d_in[7];
    const float* rW   = (const float*)d_in[8];
    const float* gW1  = (const float*)d_in[9];
    const float* gb1  = (const float*)d_in[10];
    const float* gW2  = (const float*)d_in[11];
    const float* gb2  = (const float*)d_in[12];
    const float* Wu   = (const float*)d_in[13];
    const float* bu   = (const float*)d_in[14];
    const float* Wr   = (const float*)d_in[15];
    const float* br   = (const float*)d_in[16];
    const float* Wc   = (const float*)d_in[17];
    const float* bc   = (const float*)d_in[18];
    float* out = (float*)d_out;

    char* ws = (char*)d_ws;
    float*    gn    = (float*)(ws + OFF_GN);
    float*    adjs  = (float*)(ws + OFF_ADJS);
    float*    vto   = (float*)(ws + OFF_VTO);
    unsigned* bar   = (unsigned*)(ws + OFF_BAR);
    float*    hbuf  = (float*)(ws + OFF_H);
    _Float16* c16   = (_Float16*)(ws + OFF_COMB);
    _Float16* x16   = (_Float16*)(ws + OFF_XCAT);
    _Float16* w16r  = (_Float16*)(ws + OFF_WR);
    _Float16* w16u  = (_Float16*)(ws + OFF_WU);
    _Float16* w16c  = (_Float16*)(ws + OFF_WC);

    k_gn  <<<128, 128, 0, stream>>>(vp, gW1, gb1, gW2, gb2, gn);
    k_adjs<<<128, 128, 0, stream>>>(gn, adjs);
    k_vto <<<(BB * NN + 255) / 256, 256, 0, stream>>>(mask, vto);
    k_wcvt<<<(NN * KP * DD + 255) / 256, 256, 0, stream>>>(Wr, Wu, Wc, w16r, w16u, w16c);
    k_init<<<((1048576 + 1310720 + 1310720) / 4 + 255) / 256, 256, 0, stream>>>(
        (unsigned*)(ws + OFF_H), bar);
    k_scan<<<NBLK, 256, 0, stream>>>(obs, mask, len, avg, rW, br, bu, bc,
                                     adjs, vto, hbuf, c16, x16,
                                     w16r, w16u, w16c, bar, out);
}
